// OriginalModel_gcn_2_36421322670432
// MI455X (gfx1250) — matrix operand order repaired and checked by reading, not compiled, not run on hardware
//
#include <hip/hip_runtime.h>
#include <hip/hip_bf16.h>

// ---------------- problem constants ----------------
#define BATCH   64
#define NNODES  39
#define NEDGE   1482
#define EXTRA_D 2412            // 1280 + 2 + 565 + 565 (reference comment "3412" is wrong)
#define FCIN    (64 * NNODES)   // 2496
#define FRS     24              // bf16 elems per fragment lane row (16 + 8 pad; 48B = 16B aligned)

typedef __attribute__((ext_vector_type(16))) __bf16 v16bf;
typedef __attribute__((ext_vector_type(8)))  float  v8f;
typedef __attribute__((ext_vector_type(4)))  int    v4i_t;

// ---------------- optional async global->LDS path (gfx1250) ----------------
// GLOBAL_LOAD_ASYNC_TO_LDS_B128 (ASYNCcnt-tracked). Guarded by __has_builtin so
// toolchains without the builtin compile the scalar fallback unchanged.
// Probe-derived signature: (v4i addrspace(1)* src, v4i addrspace(3)* dst,
//                           imm int offset, imm int cpol)
#if defined(__HIP_DEVICE_COMPILE__) && \
    __has_builtin(__builtin_amdgcn_global_load_async_to_lds_b128) && \
    __has_builtin(__builtin_amdgcn_s_wait_asynccnt)
#define ASYNC_LDS 1
#else
#define ASYNC_LDS 0
#endif

#if ASYNC_LDS
template <int OFF>
__device__ __forceinline__ void async_ld_b128(const void* g, void* l) {
  typedef __attribute__((address_space(1))) v4i_t* gp_t;
  typedef __attribute__((address_space(3))) v4i_t* lp_t;
  // integer round-trip for the address-space casts (compile-only environment)
  __builtin_amdgcn_global_load_async_to_lds_b128(
      (gp_t)(unsigned long long)g,
      (lp_t)(unsigned int)(unsigned long long)l,
      OFF, 0);
}
__device__ __forceinline__ void async_wait0() { __builtin_amdgcn_s_wait_asynccnt(0); }
#else
__device__ __forceinline__ void async_wait0() {}
#endif

// ---------------- fragment helpers ----------------
// LDS tiles are stored FRAGMENT-MAJOR so each lane's 16 bf16 fragment elements
// are 32 contiguous bytes -> 2x ds_load_b128 per fragment.
// A (16x32):             (m,k) -> lane = 16*(k>>4) + (m&15),     e = k&15
// B (32x16):             (k,n) -> lane = 16*(k>>4) + n,          e = k&15
__device__ __forceinline__ v16bf frag_ld(const __bf16* p) {
  union { uint4 u[2]; v16bf v; } r;
  r.u[0] = *(const uint4*)(p);
  r.u[1] = *(const uint4*)(p + 8);
  return r.v;
}

__device__ __forceinline__ v8f wmma_bf16(v16bf a, v16bf b, v8f acc) {
  return __builtin_amdgcn_wmma_f32_16x16x32_bf16(false, a, false, b,
                                                 (short)0, acc, false, false);
}

// ---------------- utility kernels ----------------
// f32 (rows,K) -> bf16 (rows,Kp) with zero pad; Kp multiple of 16 keeps every
// 32-elem K-chunk's row address 16B-aligned for async b128 loads.
__global__ void k_cast_pad(const float* __restrict__ src, __bf16* __restrict__ dst,
                           long rows, int K, int Kp) {
  long i = (long)blockIdx.x * 256 + threadIdx.x;
  if (i >= rows * Kp) return;
  int k = (int)(i % Kp);
  long r = i / Kp;
  dst[i] = (k < K) ? (__bf16)src[r * K + k] : (__bf16)0.0f;
}

__global__ void k_zero(float* __restrict__ p, long n) {
  long i = (long)blockIdx.x * 256 + threadIdx.x;
  if (i < n) p[i] = 0.0f;
}

// (B,N,W,3) f32 -> (B*N, 3, 1000) bf16
__global__ void k_pack_input(const float* __restrict__ wav,
                             __bf16* __restrict__ x0) {
  long i = (long)blockIdx.x * 256 + threadIdx.x;
  long total = (long)BATCH * NNODES * 3 * 1000;
  if (i >= total) return;
  int w = (int)(i % 1000);
  long r = i / 1000;
  int c = (int)(r % 3);
  long bn = r / 3;
  x0[i] = (__bf16)wav[(bn * 1000 + w) * 3 + c];
}

// ---------------- implicit-GEMM conv with WMMA ----------------
// xin : (B*N, Cin, Win) bf16        wt : (Cout, Kp) bf16 (logical K = Cin*125)
// out_bf : (B*N, Cout, Wout) bf16   out_f32 : (B, Cout, N, Wout) f32 (optional)
// 256 thr = 8 waves (2 M x 4 N); block tile = 32 cout x 128 spatial; 2 WMMA/wave/chunk
__global__ __launch_bounds__(256) void k_conv_wmma(
    const __bf16* __restrict__ xin, const __bf16* __restrict__ wt,
    const float* __restrict__ bias, __bf16* __restrict__ out_bf,
    float* __restrict__ out_f32, int BN, int Nn, int Cin, int Win,
    int Cout, int Wout, int K, int Kp) {
  __shared__ alignas(16) __bf16 Afr[2 * 32 * FRS];   // [msub][lane][e]
  __shared__ alignas(16) __bf16 Bfr[8 * 32 * FRS];   // [nsub][lane][e]
  const int t = threadIdx.x;
  const int wave = t >> 5, wm = wave >> 2, wn = wave & 3;
  const int lane = t & 31, l = lane & 15, hi = lane >> 4;
  const int s0 = blockIdx.x * 128;
  const int m0 = blockIdx.y * 32;
  const int total = BN * Wout;

  // B-filler geometry: this thread owns spatial column `bcol`, k-range [bkb, bkb+16)
  const int bcol = t & 127;
  const int bkb  = (t >> 7) << 4;                    // 0 or 16
  const int s_my = s0 + bcol;
  int bn_my = 0, w_my = 0;
  if (s_my < total) { bn_my = s_my / Wout; w_my = s_my - bn_my * Wout; }
  __bf16* bdst = &Bfr[(((bcol >> 4) * 32) + (bkb ? 16 : 0) + (bcol & 15)) * FRS];

  // A-filler geometry (threads 0..127): one 8-elem run each
  const int amm = t & 31, akgi = t >> 5;             // akgi 0..3 -> kkbase 0,8,16,24
  const int akkb = akgi * 8;
  __bf16* adst = &Afr[((((amm >> 4) * 32) + ((akgi >> 1) << 4) + (amm & 15)) * FRS) +
                      ((akgi & 1) ? 8 : 0)];

  v8f acc0 = {}, acc1 = {};
  for (int k0 = 0; k0 < K; k0 += 32) {
    if (k0 + 32 < K)  // global_prefetch_b8 on next weight chunk
      __builtin_prefetch(&wt[(size_t)(m0 + lane) * Kp + k0 + 32], 0, 0);
    // ---- A tile (weights) ----
    if (t < 128) {
      const __bf16* gsrc = &wt[(size_t)(m0 + amm) * Kp + k0 + akkb];
#if ASYNC_LDS
      if (k0 + 32 <= K) {
        async_ld_b128<0>(gsrc, adst);                // 16B run, ASYNCcnt-tracked
      } else
#endif
      {
        __bf16 tmp[8];
#pragma unroll
        for (int j = 0; j < 8; ++j)
          tmp[j] = (k0 + akkb + j < K) ? gsrc[j] : (__bf16)0.0f;
        __builtin_memcpy(adst, tmp, 16);
      }
    }
    // ---- B tile (im2col gather): 16-elem run per thread (2x ds_store_b128) ----
    {
      __bf16 tmp[16];
#pragma unroll
      for (int e = 0; e < 16; ++e) {
        int kg = k0 + bkb + e;
        __bf16 v = (__bf16)0.0f;
        if (kg < K && s_my < total) {
          int c = kg / 125, kw = kg - c * 125;
          v = xin[((size_t)bn_my * Cin + c) * Win + 2 * w_my + kw];
        }
        tmp[e] = v;
      }
      __builtin_memcpy(bdst, tmp, 32);
    }
    async_wait0();
    __syncthreads();
    v16bf af  = frag_ld(&Afr[(wm * 32 + lane) * FRS]);
    v16bf bf0 = frag_ld(&Bfr[((wn * 2 + 0) * 32 + lane) * FRS]);
    v16bf bf1 = frag_ld(&Bfr[((wn * 2 + 1) * 32 + lane) * FRS]);
    acc0 = wmma_bf16(af, bf0, acc0);
    acc1 = wmma_bf16(af, bf1, acc1);
    __syncthreads();
  }
  // ---- epilogue: bias + relu; C layout per ISA (M = vgpr + 8*hi, N = lane&15)
#pragma unroll
  for (int j = 0; j < 2; ++j) {
    const v8f& acc = j ? acc1 : acc0;
    int s = s0 + wn * 32 + j * 16 + l;
    if (s < total) {
      int bn = s / Wout, w = s - bn * Wout;
      int b = bn / Nn, n = bn - b * Nn;
#pragma unroll
      for (int i = 0; i < 8; ++i) {
        int m = m0 + wm * 16 + i + hi * 8;
        float v = acc[i] + bias[m];
        v = v > 0.0f ? v : 0.0f;               // all three convs are conv+relu
        if (out_bf)  out_bf[((size_t)bn * Cout + m) * Wout + w] = (__bf16)v;
        if (out_f32) out_f32[(((size_t)b * Cout + m) * Nn + n) * Wout + w] = v;
      }
    }
  }
}

// ---------------- generic dense GEMM with WMMA ----------------
// act : (M,K) f32 (cast to bf16 on LDS fill)   wbf : (O,Kp) bf16
// out[m*O+o] = act(m,:) . wbf(o,:) + bias[o]  (optional relu)
// block tile: 32 M x 128 O
__global__ __launch_bounds__(256) void k_gemm_wmma(
    const float* __restrict__ act, const __bf16* __restrict__ wbf,
    const float* __restrict__ bias, float* __restrict__ out,
    int M, int K, int Kp, int O, int relu) {
  __shared__ alignas(16) __bf16 Afr[2 * 32 * FRS];
  __shared__ alignas(16) __bf16 Bfr[8 * 32 * FRS];
  const int t = threadIdx.x;
  const int wave = t >> 5, wm = wave >> 2, wn = wave & 3;
  const int lane = t & 31, l = lane & 15, hi = lane >> 4;
  const int m0 = blockIdx.y * 32;
  const int o0 = blockIdx.x * 128;

  const int bcol = t & 127;                      // output-channel column
  const int bkb  = (t >> 7) << 4;
  const int o_my = o0 + bcol;
  __bf16* bdst = &Bfr[(((bcol >> 4) * 32) + (bkb ? 16 : 0) + (bcol & 15)) * FRS];

  const int amm = t & 31, akgi = t >> 5;
  const int akkb = akgi * 8;
  __bf16* adst = &Afr[((((amm >> 4) * 32) + ((akgi >> 1) << 4) + (amm & 15)) * FRS) +
                      ((akgi & 1) ? 8 : 0)];

  v8f acc0 = {}, acc1 = {};
  for (int k0 = 0; k0 < K; k0 += 32) {
    if (k0 + 32 < K && o_my < O)
      __builtin_prefetch(&wbf[(size_t)o_my * Kp + k0 + 32], 0, 0);
    // ---- A tile: activations f32 -> bf16, 8-elem runs ----
    if (t < 128) {
      int m = m0 + amm;
      __bf16 tmp[8];
#pragma unroll
      for (int j = 0; j < 8; ++j) {
        int kg = k0 + akkb + j;
        float v = (m < M && kg < K) ? act[(size_t)m * K + kg] : 0.0f;
        tmp[j] = (__bf16)v;
      }
      __builtin_memcpy(adst, tmp, 16);
    }
    // ---- B tile: weight rows, contiguous 32B runs ----
    {
      const __bf16* gsrc = &wbf[(size_t)o_my * Kp + k0 + bkb];
#if ASYNC_LDS
      if (o_my < O && k0 + 32 <= K) {
        async_ld_b128<0>(gsrc, bdst);              // two b128 asyncs cover 32B
        async_ld_b128<16>(gsrc, bdst);             // imm offset applies to both addrs
      } else
#endif
      {
        __bf16 tmp[16];
#pragma unroll
        for (int e = 0; e < 16; ++e) {
          int kg = k0 + bkb + e;
          tmp[e] = (o_my < O && kg < K) ? gsrc[e] : (__bf16)0.0f;
        }
        __builtin_memcpy(bdst, tmp, 32);
      }
    }
    async_wait0();
    __syncthreads();
    v16bf af  = frag_ld(&Afr[(wm * 32 + lane) * FRS]);
    v16bf bf0 = frag_ld(&Bfr[((wn * 2 + 0) * 32 + lane) * FRS]);
    v16bf bf1 = frag_ld(&Bfr[((wn * 2 + 1) * 32 + lane) * FRS]);
    acc0 = wmma_bf16(af, bf0, acc0);
    acc1 = wmma_bf16(af, bf1, acc1);
    __syncthreads();
  }
#pragma unroll
  for (int j = 0; j < 2; ++j) {
    const v8f& acc = j ? acc1 : acc0;
    int o = o0 + wn * 32 + j * 16 + l;
    if (o < O) {
#pragma unroll
      for (int i = 0; i < 8; ++i) {
        int m = m0 + wm * 16 + i + hi * 8;
        if (m < M) {
          float v = acc[i] + bias[o];
          if (relu) v = v > 0.0f ? v : 0.0f;
          out[(size_t)m * O + o] = v;
        }
      }
    }
  }
}

// ---------------- GCN helpers (tiny, scalar) ----------------
__global__ void k_gcn_prep(const int* __restrict__ g, const float* __restrict__ ew,
                           float* __restrict__ deg, float* __restrict__ norm) {
  const int* src = g;
  const int* tgt = g + NEDGE;
  int t = threadIdx.x;
  for (int n = t; n < NNODES; n += 256) deg[n] = 1.0f;   // self-loop weight
  __syncthreads();
  for (int e = t; e < NEDGE; e += 256) atomicAdd(&deg[tgt[e]], ew[e]);
  __threadfence();
  __syncthreads();
  for (int e = t; e < NEDGE + NNODES; e += 256) {
    int s_, t_; float w_;
    if (e < NEDGE) { s_ = src[e]; t_ = tgt[e]; w_ = ew[e]; }
    else           { s_ = t_ = e - NEDGE;      w_ = 1.0f; }
    float ds = deg[s_], dt = deg[t_];
    float is = ds > 0.0f ? rsqrtf(ds) : 0.0f;
    float it = dt > 0.0f ? rsqrtf(dt) : 0.0f;
    norm[e] = is * w_ * it;
  }
}

// out(rows,O) = x(rows,K) @ w(K,O)
__global__ void k_matmul_scalar(const float* __restrict__ x,
                                const float* __restrict__ w,
                                float* __restrict__ out, int rows, int K, int O) {
  long i = (long)blockIdx.x * 256 + threadIdx.x;
  if (i >= (long)rows * O) return;
  int o = (int)(i % O);
  long r = i / O;
  const float* xr = x + r * K;
  float s = 0.0f;
  for (int k = 0; k < K; ++k) s += xr[k] * w[(size_t)k * O + o];
  out[r * O + o] = s;
}

// out[b,tgt,o] += norm[e] * xw[b,src,o]   (self-loops appended at e >= NEDGE)
__global__ void k_gcn_scatter(const float* __restrict__ xw, const int* __restrict__ g,
                              const float* __restrict__ norm, float* __restrict__ out) {
  long i = (long)blockIdx.x * 256 + threadIdx.x;
  long total = (long)BATCH * (NEDGE + NNODES) * 64;
  if (i >= total) return;
  int o = (int)(i & 63);
  long r = i >> 6;
  int e = (int)(r % (NEDGE + NNODES));
  int b = (int)(r / (NEDGE + NNODES));
  int s, t;
  if (e < NEDGE) { s = g[e]; t = g[NEDGE + e]; }
  else           { s = t = e - NEDGE; }
  atomicAdd(&out[((long)b * NNODES + t) * 64 + o],
            norm[e] * xw[((long)b * NNODES + s) * 64 + o]);
}

__global__ void k_bias_act(float* __restrict__ buf, const float* __restrict__ bias,
                           long rows, int O, int mode) {  // 1=relu, 2=tanh
  long i = (long)blockIdx.x * 256 + threadIdx.x;
  if (i >= rows * O) return;
  int o = (int)(i % O);
  float v = buf[i] + bias[o];
  if (mode == 1)      v = fmaxf(v, 0.0f);
  else if (mode == 2) v = tanhf(v);
  buf[i] = v;
}

__global__ void k_concat_extra(const float* __restrict__ fc1o,
                               const float* __restrict__ arr,
                               const float* __restrict__ mx,
                               const float* __restrict__ dist,
                               float* __restrict__ xe) {
  int i = blockIdx.x * 256 + threadIdx.x;
  if (i >= BATCH * EXTRA_D) return;
  int b = i / EXTRA_D, c = i % EXTRA_D;
  float v;
  if (c < 1280)      v = fc1o[b * 1280 + c];
  else if (c < 1282) v = arr[b * 2 + (c - 1280)];
  else if (c < 1847) v = mx[b * 565 + (c - 1282)];
  else               v = dist[b * 565 + (c - 1847)];
  xe[i] = v;
}

// ---------------- host orchestration ----------------
extern "C" void kernel_launch(void* const* d_in, const int* in_sizes, int n_in,
                              void* d_out, int out_size, void* d_ws, size_t ws_size,
                              hipStream_t stream) {
  (void)in_sizes; (void)n_in; (void)out_size; (void)ws_size;
  const float* wav   = (const float*)d_in[0];
  const int*   graph = (const int*)d_in[1];
  const float* ew    = (const float*)d_in[2];
  const float* arr   = (const float*)d_in[3];
  const float* mxv   = (const float*)d_in[4];
  const float* dist  = (const float*)d_in[5];

  const int BN = BATCH * NNODES;     // 2496
  auto kpad = [](int K) { return (K + 15) & ~15; };
  const int KP1 = kpad(375);         // 384
  const int KP2 = 4000, KP3 = 8000, KPF1 = kpad(FCIN), KPE = kpad(EXTRA_D); // 2496, 2416

  char* ws = (char*)d_ws;
  size_t off = 0;
  auto take = [&](size_t bytes) {
    size_t o = off;
    off = (off + bytes + 255) & ~(size_t)255;
    return o;
  };
  __bf16* x0   = (__bf16*)(ws + take((size_t)BN * 3 * 1000 * 2));
  __bf16* wt1  = (__bf16*)(ws + take((size_t)32 * KP1 * 2));
  __bf16* wt2  = (__bf16*)(ws + take((size_t)64 * KP2 * 2));
  __bf16* wt3  = (__bf16*)(ws + take((size_t)128 * KP3 * 2));
  __bf16* y1   = (__bf16*)(ws + take((size_t)BN * 32 * 438 * 2));
  __bf16* y2   = (__bf16*)(ws + take((size_t)BN * 64 * 157 * 2));
  float*  y3   = (float*)(ws + take((size_t)BATCH * 128 * NNODES * 17 * 4));
  float*  bufA = (float*)(ws + take((size_t)BN * 64 * 4));
  float*  bufB = (float*)(ws + take((size_t)BN * 64 * 4));
  float*  deg  = (float*)(ws + take((size_t)NNODES * 4));
  float*  nrm  = (float*)(ws + take((size_t)(NEDGE + NNODES) * 4));
  __bf16* fc1w = (__bf16*)(ws + take((size_t)1280 * KPF1 * 2));
  float*  fc1o = (float*)(ws + take((size_t)BATCH * 1280 * 4));
  float*  xe   = (float*)(ws + take((size_t)BATCH * EXTRA_D * 4));
  __bf16* fcew = (__bf16*)(ws + take((size_t)EXTRA_D * KPE * 2));
  float*  fceo = (float*)(ws + take((size_t)BATCH * EXTRA_D * 4));
  __bf16* hw   = (__bf16*)(ws + take((size_t)5 * NNODES * KPE * 2));

  auto blks = [](long n) { return (unsigned)((n + 255) / 256); };

  // 1. pack input (B,N,W,3) -> (B*N,3,1000) bf16
  k_pack_input<<<blks((long)BN * 3000), 256, 0, stream>>>(wav, x0);

  // 2. weight casts f32 -> bf16 (row-padded to 16-elem multiples)
  k_cast_pad<<<blks(32L * KP1), 256, 0, stream>>>((const float*)d_in[6],  wt1, 32, 375, KP1);
  k_cast_pad<<<blks(64L * KP2), 256, 0, stream>>>((const float*)d_in[8],  wt2, 64, 4000, KP2);
  k_cast_pad<<<blks(128L * KP3), 256, 0, stream>>>((const float*)d_in[10], wt3, 128, 8000, KP3);
  k_cast_pad<<<blks(1280L * KPF1), 256, 0, stream>>>((const float*)d_in[16], fc1w, 1280, FCIN, KPF1);
  k_cast_pad<<<blks((long)EXTRA_D * KPE), 256, 0, stream>>>((const float*)d_in[18], fcew,
                                                            EXTRA_D, EXTRA_D, KPE);
  for (int h = 0; h < 5; ++h)
    k_cast_pad<<<blks((long)NNODES * KPE), 256, 0, stream>>>(
        (const float*)d_in[20 + 2 * h], hw + (size_t)h * NNODES * KPE,
        NNODES, EXTRA_D, KPE);

  // 3. conv stack (implicit GEMM, bf16 WMMA)
  {
    long tot = (long)BN * 438;
    dim3 g((unsigned)((tot + 127) / 128), 1);
    k_conv_wmma<<<g, 256, 0, stream>>>(x0, wt1, (const float*)d_in[7], y1, nullptr,
                                       BN, NNODES, 3, 1000, 32, 438, 375, KP1);
  }
  {
    long tot = (long)BN * 157;
    dim3 g((unsigned)((tot + 127) / 128), 2);
    k_conv_wmma<<<g, 256, 0, stream>>>(y1, wt2, (const float*)d_in[9], y2, nullptr,
                                       BN, NNODES, 32, 438, 64, 157, 4000, KP2);
  }
  {
    long tot = (long)BN * 17;
    dim3 g((unsigned)((tot + 127) / 128), 4);
    k_conv_wmma<<<g, 256, 0, stream>>>(y2, wt3, (const float*)d_in[11], nullptr, y3,
                                       BN, NNODES, 64, 157, 128, 17, 8000, KP3);
  }

  // 4. GCN: y3 flat (B,128,N,17) row-major IS the "raw reshape" (B,N,2176)
  k_gcn_prep<<<1, 256, 0, stream>>>(graph, ew, deg, nrm);
  k_matmul_scalar<<<blks((long)BN * 64), 256, 0, stream>>>(y3, (const float*)d_in[12],
                                                           bufA, BN, 2176, 64);
  k_zero<<<blks((long)BN * 64), 256, 0, stream>>>(bufB, (long)BN * 64);
  k_gcn_scatter<<<blks((long)BATCH * (NEDGE + NNODES) * 64), 256, 0, stream>>>(bufA, graph, nrm, bufB);
  k_bias_act<<<blks((long)BN * 64), 256, 0, stream>>>(bufB, (const float*)d_in[13], BN, 64, 1);
  k_matmul_scalar<<<blks((long)BN * 64), 256, 0, stream>>>(bufB, (const float*)d_in[14],
                                                           bufA, BN, 64, 64);
  k_zero<<<blks((long)BN * 64), 256, 0, stream>>>(bufB, (long)BN * 64);
  k_gcn_scatter<<<blks((long)BATCH * (NEDGE + NNODES) * 64), 256, 0, stream>>>(bufA, graph, nrm, bufB);
  k_bias_act<<<blks((long)BN * 64), 256, 0, stream>>>(bufB, (const float*)d_in[15], BN, 64, 2);

  // 5. fc1: (64,2496) @ (2496->1280) + relu  (bufB flat is (B, 64*N))
  {
    dim3 g((1280 + 127) / 128, (BATCH + 31) / 32);
    k_gemm_wmma<<<g, 256, 0, stream>>>(bufB, fc1w, (const float*)d_in[17], fc1o,
                                       BATCH, FCIN, KPF1, 1280, 1);
  }
  // 6. concat extras -> (64, 2412)
  k_concat_extra<<<blks((long)BATCH * EXTRA_D), 256, 0, stream>>>(fc1o, arr, mxv, dist, xe);
  // 7. fce: (64,2412) @ (2412->2412) + relu
  {
    dim3 g((EXTRA_D + 127) / 128, (BATCH + 31) / 32);
    k_gemm_wmma<<<g, 256, 0, stream>>>(xe, fcew, (const float*)d_in[19], fceo,
                                       BATCH, EXTRA_D, KPE, EXTRA_D, 1);
  }
  // 8. five heads -> d_out, concatenated (B,39) each
  for (int h = 0; h < 5; ++h) {
    dim3 g((NNODES + 127) / 128, (BATCH + 31) / 32);
    k_gemm_wmma<<<g, 256, 0, stream>>>(fceo, hw + (size_t)h * NNODES * KPE,
                                       (const float*)d_in[21 + 2 * h],
                                       (float*)d_out + (size_t)h * BATCH * NNODES,
                                       BATCH, EXTRA_D, KPE, NNODES, 0);
  }
}
